// DynamicGraphConv_44573170598112
// MI455X (gfx1250) — matrix operand order repaired and checked by reading, not compiled, not run on hardware
//
#include <hip/hip_runtime.h>

// Problem constants (match reference)
#define BB   16
#define CC   256
#define LL   1024
#define HIDN 128
#define NHD  4
#define HDD  32

typedef unsigned short ushort_t;
typedef __attribute__((ext_vector_type(16))) __bf16 v16bf;
typedef __attribute__((ext_vector_type(8)))  __bf16 v8bf;
typedef __attribute__((ext_vector_type(8)))  float  v8f;

// ---------- helpers -------------------------------------------------------

// float -> bf16 bits, round-to-nearest-even (pure integer math)
static __device__ __forceinline__ ushort_t f2bfu(float f) {
  unsigned u = __builtin_bit_cast(unsigned, f);
  u += 0x7fffu + ((u >> 16) & 1u);
  return (ushort_t)(u >> 16);
}

// A fragment: 16x32 bf16 tile, row-major memory, lda multiple of 8.
// ISA layout: row M = lane&15; lanes 0-15 hold K {0..7,16..23},
// lanes 16-31 hold K {8..15,24..31}.  Two 16-byte loads per lane.
static __device__ __forceinline__ v16bf load_a_bf(const ushort_t* __restrict__ A,
                                                  int lda) {
  int lane = threadIdx.x & 31, row = lane & 15;
  int kb = (lane < 16) ? 0 : 8;
  const ushort_t* p = A + (size_t)row * lda + kb;
  v8bf lo = *(const v8bf*)p;
  v8bf hi = *(const v8bf*)(p + 16);
  return __builtin_shufflevector(lo, hi, 0, 1, 2, 3, 4, 5, 6, 7,
                                 8, 9, 10, 11, 12, 13, 14, 15);
}

// B fragment: 32x16 bf16 tile whose SOURCE is stored N-major (transposed):
// B[k,n] = Bt[n*ldbt + k].  col N = lane&15; lanes 0-15 K {0..7,16..23},
// lanes 16-31 K {8..15,24..31}.  Two 16-byte loads per lane.
static __device__ __forceinline__ v16bf load_bt_bf(const ushort_t* __restrict__ Bt,
                                                   int ldbt) {
  int lane = threadIdx.x & 31, col = lane & 15;
  int kb = (lane < 16) ? 0 : 8;
  const ushort_t* p = Bt + (size_t)col * ldbt + kb;
  v8bf lo = *(const v8bf*)p;
  v8bf hi = *(const v8bf*)(p + 16);
  return __builtin_shufflevector(lo, hi, 0, 1, 2, 3, 4, 5, 6, 7,
                                 8, 9, 10, 11, 12, 13, 14, 15);
}

static __device__ __forceinline__ v8f wmma_bf16(v16bf a, v16bf b, v8f c) {
  return __builtin_amdgcn_wmma_f32_16x16x32_bf16(false, a, false, b, (short)0, c,
                                                 false, false);
}

// ---------- conversion pre-passes ----------------------------------------

// src [K,N] f32 row-major  ->  dst [N,K] bf16 (N-major for B fragments)
__global__ void cvt_transpose_kernel(const float* __restrict__ src,
                                     ushort_t* __restrict__ dst, int K, int N) {
  int id = blockIdx.x * 256 + threadIdx.x;
  if (id >= K * N) return;
  int n = id / K, k = id - n * K;
  dst[id] = f2bfu(src[(size_t)k * N + n]);
}

// x [B,C,L] f32  ->  xt [B,L,C] bf16 (row-major rows of length C)
__global__ void cvt_x_kernel(const float* __restrict__ x,
                             ushort_t* __restrict__ xt) {
  int id = blockIdx.x * 256 + threadIdx.x;   // over B*L*C
  int c = id & (CC - 1);
  int l = (id >> 8) & (LL - 1);
  int b = id >> 18;
  xt[id] = f2bfu(x[((size_t)b * CC + c) * LL + l]);
}

// adj f32 -> bf16, 8 elements per thread (b128 in, b128 out)
__global__ void cvt_adj_kernel(const float* __restrict__ adj,
                               ushort_t* __restrict__ dst) {
  size_t base = ((size_t)blockIdx.x * 256 + threadIdx.x) * 8;
  const float4* s = (const float4*)(adj + base);
  float4 f0 = s[0], f1 = s[1];
  uint4 u;
  u.x = (unsigned)f2bfu(f0.x) | ((unsigned)f2bfu(f0.y) << 16);
  u.y = (unsigned)f2bfu(f0.z) | ((unsigned)f2bfu(f0.w) << 16);
  u.z = (unsigned)f2bfu(f1.x) | ((unsigned)f2bfu(f1.y) << 16);
  u.w = (unsigned)f2bfu(f1.z) | ((unsigned)f2bfu(f1.w) << 16);
  *(uint4*)(dst + base) = u;
}

// ---------- kernel 1: QKV projection (bf16 in, bf16 out) ------------------
// grid (LL/16, BB, 3) block 256 (8 waves); wave w -> N-tile w of HID
__global__ void qkv_kernel(const ushort_t* __restrict__ xt,
                           const ushort_t* __restrict__ wqt,
                           const ushort_t* __restrict__ wkt,
                           const ushort_t* __restrict__ wvt,
                           const float* __restrict__ bq,
                           const float* __restrict__ bk,
                           const float* __restrict__ bv,
                           ushort_t* __restrict__ q, ushort_t* __restrict__ k,
                           ushort_t* __restrict__ vt) {
  int b = blockIdx.y, l0 = blockIdx.x * 16, z = blockIdx.z;
  int wave = threadIdx.x >> 5, lane = threadIdx.x & 31;
  int n0 = wave * 16;
  const ushort_t* wt  = (z == 0) ? wqt : (z == 1) ? wkt : wvt;
  const float*    bia = (z == 0) ? bq : (z == 1) ? bk : bv;

  const ushort_t* xb = xt + ((size_t)b * LL + l0) * CC;
  v8f acc = {};
#pragma unroll
  for (int k0 = 0; k0 < CC; k0 += 32) {
    v16bf a  = load_a_bf(xb + k0, CC);
    v16bf bf = load_bt_bf(wt + (size_t)n0 * CC + k0, CC);
    acc = wmma_bf16(a, bf, acc);
  }
  int col = lane & 15, mofs = (lane < 16) ? 0 : 8;
  int ng = n0 + col, h = ng >> 5, d = ng & 31;
  float bz = bia[ng];
#pragma unroll
  for (int i = 0; i < 8; ++i) {
    int l = l0 + mofs + i;
    ushort_t r = f2bfu(acc[i] + bz);
    if (z == 0)      q[(((size_t)b * NHD + h) * LL + l) * HDD + d] = r;
    else if (z == 1) k[(((size_t)b * NHD + h) * LL + l) * HDD + d] = r;
    else             vt[(((size_t)b * NHD + h) * HDD + d) * LL + l] = r; // d-major
  }
}

// ---------- kernel 2: fused attention -------------------------------------
// grid (LL/32, BB) block 64 (2 waves); wave owns 16 query rows.
// LDS/wave: 64KB f32 scores + 32KB bf16 probs -> 192KB/block (of 320KB WGP).
__global__ void attn_kernel(const ushort_t* __restrict__ q,
                            const ushort_t* __restrict__ k,
                            const ushort_t* __restrict__ vt,
                            const float* __restrict__ dm,
                            const float* __restrict__ alpha_p,
                            ushort_t* __restrict__ o, float* __restrict__ adj) {
  extern __shared__ float smem[];   // [2*16*1024 f32][2*16*1024 bf16]
  int b = blockIdx.y;
  int wave = threadIdx.x >> 5, lane = threadIdx.x & 31;
  int qr0 = blockIdx.x * 32 + wave * 16;
  float alpha = alpha_p[0];
  const float scale = 0.17677669529663687f;   // 1/sqrt(32)
  float*    sc = smem + (size_t)wave * 16 * LL;
  ushort_t* pb = (ushort_t*)(smem + 2 * 16 * LL) + (size_t)wave * 16 * LL;
  int col = lane & 15, mofs = (lane < 16) ? 0 : 8;
  float* adjb = adj + (size_t)b * LL * LL;

  for (int h = 0; h < NHD; ++h) {
    const ushort_t* qh = q + ((size_t)b * NHD + h) * LL * HDD;
    const ushort_t* kh = k + ((size_t)b * NHD + h) * LL * HDD;
    const ushort_t* vh = vt + ((size_t)b * NHD + h) * HDD * LL;  // [HD][L]

    // ---- scores = (Q K^T) * scale * dist_inf -> LDS f32 --------------
    v16bf aq = load_a_bf(qh + (size_t)qr0 * HDD, HDD);   // K-depth 32 == HD
    for (int kt = 0; kt < LL / 16; ++kt) {
      v16bf bk = load_bt_bf(kh + (size_t)kt * 16 * HDD, HDD);
      v8f cc = {};
      cc = wmma_bf16(aq, bk, cc);
#pragma unroll
      for (int i = 0; i < 8; ++i) {
        int m = mofs + i;
        int gr = qr0 + m, gc = kt * 16 + col;
        float dinf = 1.0f / (alpha * dm[(size_t)gr * LL + gc] + 1.0f);
        sc[(size_t)m * LL + gc] = cc[i] * scale * dinf;
      }
    }

    // ---- row softmax (wave32 shuffles); probs -> bf16 LDS; adj accum --
    for (int r = 0; r < 16; ++r) {
      float mx = -3.0e38f;
      for (int c = lane; c < LL; c += 32) mx = fmaxf(mx, sc[(size_t)r * LL + c]);
#pragma unroll
      for (int s = 16; s > 0; s >>= 1) mx = fmaxf(mx, __shfl_xor(mx, s));
      float sum = 0.f;
      for (int c = lane; c < LL; c += 32) {
        float e = __expf(sc[(size_t)r * LL + c] - mx);
        sc[(size_t)r * LL + c] = e;
        sum += e;
      }
#pragma unroll
      for (int s = 16; s > 0; s >>= 1) sum += __shfl_xor(sum, s);
      float inv = 1.0f / sum;
      float* arow = adjb + (size_t)(qr0 + r) * LL;
      for (int c = lane; c < LL; c += 32) {
        float p = sc[(size_t)r * LL + c] * inv;
        pb[(size_t)r * LL + c] = f2bfu(p);
        float av = 0.25f * p;                       // mean over NH=4 heads
        if (h == 0) arow[c] = av; else arow[c] += av;
      }
    }

    // ---- o = attn @ V : A from bf16 LDS (ds_load_b128), B from v_t ----
    v8f c0 = {}, c1 = {};
    int row = lane & 15, kbA = (lane < 16) ? 0 : 8;
    for (int kt = 0; kt < LL / 32; ++kt) {
      const ushort_t* pp = pb + (size_t)row * LL + kt * 32 + kbA;
      v8bf lo = *(const v8bf*)pp;
      v8bf hi = *(const v8bf*)(pp + 16);
      v16bf ap = __builtin_shufflevector(lo, hi, 0, 1, 2, 3, 4, 5, 6, 7,
                                         8, 9, 10, 11, 12, 13, 14, 15);
      v16bf b0 = load_bt_bf(vh + kt * 32, LL);              // d = 0..15
      v16bf b1 = load_bt_bf(vh + (size_t)16 * LL + kt * 32, LL); // d = 16..31
      c0 = wmma_bf16(ap, b0, c0);
      c1 = wmma_bf16(ap, b1, c1);
    }
#pragma unroll
    for (int i = 0; i < 8; ++i) {
      int l = qr0 + mofs + i;
      ushort_t* orow = o + ((size_t)b * LL + l) * HIDN + h * HDD;
      orow[col]      = f2bfu(c0[i]);
      orow[col + 16] = f2bfu(c1[i]);
    }
  }
}

// ---------- kernel 3: h_attn_t = o @ Wo + bo -> stored C-major bf16 -------
// grid (BB*LL/16, 2) block 256; wave w -> N-tile blockIdx.y*8 + w
__global__ void owo_kernel(const ushort_t* __restrict__ o,
                           const ushort_t* __restrict__ wot,
                           const float* __restrict__ bo,
                           ushort_t* __restrict__ hat) {
  int m0 = blockIdx.x * 16;
  int wave = threadIdx.x >> 5, lane = threadIdx.x & 31;
  int n0 = (blockIdx.y * 8 + wave) * 16;
  v8f acc = {};
#pragma unroll
  for (int k0 = 0; k0 < HIDN; k0 += 32) {
    v16bf a  = load_a_bf(o + (size_t)m0 * HIDN + k0, HIDN);
    v16bf bf = load_bt_bf(wot + (size_t)n0 * HIDN + k0, HIDN);
    acc = wmma_bf16(a, bf, acc);
  }
  int col = lane & 15, mofs = (lane < 16) ? 0 : 8;
  int bidx = m0 / LL, l0 = m0 - bidx * LL;
  int c = n0 + col;
  float bz = bo[c];
  ushort_t* hrow = hat + ((size_t)bidx * CC + c) * LL + l0 + mofs;
#pragma unroll
  for (int i = 0; i < 8; ++i) hrow[i] = f2bfu(acc[i] + bz);   // contiguous in L
}

// ---------- kernel 4: support = adj @ h_attn_t (both bf16) ----------------
// grid (LL/16, BB) block 256; wave w -> N-tiles w and w+8
__global__ void gcn_support_kernel(const ushort_t* __restrict__ adjbf,
                                   const ushort_t* __restrict__ hat,
                                   ushort_t* __restrict__ sup) {
  int b = blockIdx.y, m0 = blockIdx.x * 16;
  int wave = threadIdx.x >> 5, lane = threadIdx.x & 31;
  const ushort_t* adjb = adjbf + (size_t)b * LL * LL + (size_t)m0 * LL;
  const ushort_t* hb   = hat + (size_t)b * CC * LL;
  int n0 = wave * 16, n1 = (wave + 8) * 16;
  v8f acc0 = {}, acc1 = {};
  for (int k0 = 0; k0 < LL; k0 += 32) {
    __builtin_prefetch(adjb + k0 + 256, 0, 1);
    v16bf a  = load_a_bf(adjb + k0, LL);
    v16bf b0 = load_bt_bf(hb + (size_t)n0 * LL + k0, LL);
    v16bf b1 = load_bt_bf(hb + (size_t)n1 * LL + k0, LL);
    acc0 = wmma_bf16(a, b0, acc0);
    acc1 = wmma_bf16(a, b1, acc1);
  }
  int col = lane & 15, mofs = (lane < 16) ? 0 : 8;
  ushort_t* supb = sup + (size_t)b * LL * CC;
#pragma unroll
  for (int i = 0; i < 8; ++i) {
    supb[(size_t)(m0 + mofs + i) * CC + n0 + col] = f2bfu(acc0[i]);
    supb[(size_t)(m0 + mofs + i) * CC + n1 + col] = f2bfu(acc1[i]);
  }
}

// ---------- kernel 5: out = (support @ W + bias)^T + x (f32 out) ----------
// grid (LL/16, BB) block 256; wave w -> N-tiles w and w+8
__global__ void out_kernel(const ushort_t* __restrict__ sup,
                           const ushort_t* __restrict__ wgtt,
                           const float* __restrict__ bias,
                           const float* __restrict__ x,
                           float* __restrict__ out) {
  int b = blockIdx.y, m0 = blockIdx.x * 16;
  int wave = threadIdx.x >> 5, lane = threadIdx.x & 31;
  const ushort_t* supb = sup + ((size_t)b * LL + m0) * CC;
  int n0 = wave * 16, n1 = (wave + 8) * 16;
  v8f acc0 = {}, acc1 = {};
#pragma unroll
  for (int k0 = 0; k0 < CC; k0 += 32) {
    v16bf a  = load_a_bf(supb + k0, CC);
    v16bf b0 = load_bt_bf(wgtt + (size_t)n0 * CC + k0, CC);
    v16bf b1 = load_bt_bf(wgtt + (size_t)n1 * CC + k0, CC);
    acc0 = wmma_bf16(a, b0, acc0);
    acc1 = wmma_bf16(a, b1, acc1);
  }
  int col = lane & 15, mofs = (lane < 16) ? 0 : 8;
  const float* xb = x + (size_t)b * CC * LL;
  float* outb = out + (size_t)b * CC * LL;
  int c0 = n0 + col, c1 = n1 + col;
  float bi0 = bias[c0], bi1 = bias[c1];
#pragma unroll
  for (int i = 0; i < 8; ++i) {
    int l = m0 + mofs + i;
    outb[(size_t)c0 * LL + l] = acc0[i] + bi0 + xb[(size_t)c0 * LL + l];
    outb[(size_t)c1 * LL + l] = acc1[i] + bi1 + xb[(size_t)c1 * LL + l];
  }
}

// ---------- host launcher -------------------------------------------------
extern "C" void kernel_launch(void* const* d_in, const int* in_sizes, int n_in,
                              void* d_out, int out_size, void* d_ws, size_t ws_size,
                              hipStream_t stream) {
  const float* x      = (const float*)d_in[0];
  const float* dm     = (const float*)d_in[1];
  const float* alpha  = (const float*)d_in[2];
  const float* wq     = (const float*)d_in[3];
  const float* bq     = (const float*)d_in[4];
  const float* wk     = (const float*)d_in[5];
  const float* bk     = (const float*)d_in[6];
  const float* wv     = (const float*)d_in[7];
  const float* bv     = (const float*)d_in[8];
  const float* wo     = (const float*)d_in[9];
  const float* bo     = (const float*)d_in[10];
  const float* weight = (const float*)d_in[11];
  const float* bias   = (const float*)d_in[12];

  float* out = (float*)d_out;                       // [B,C,L] f32
  float* adj = out + (size_t)BB * CC * LL;          // [B,L,L] f32

  // bf16 scratch (ushort element offsets)
  ushort_t* ws = (ushort_t*)d_ws;
  ushort_t* WQT = ws;                               // [HID,C]   32768
  ushort_t* WKT = ws + 32768;
  ushort_t* WVT = ws + 65536;
  ushort_t* WOT = ws + 98304;                       // [C,HID]   32768
  ushort_t* WGT = ws + 131072;                      // [C,C]     65536
  ushort_t* XT  = ws + 262144;                      // [B,L,C]   4M
  ushort_t* Q   = ws + 4456448;                     // [B,H,L,HD] 2M
  ushort_t* K   = ws + 6553600;                     // [B,H,L,HD] 2M
  ushort_t* VT  = ws + 8650752;                     // [B,H,HD,L] 2M
  ushort_t* O   = ws + 10747904;                    // [B,L,HID]  2M
  ushort_t* HAT = ws + 12845056;                    // [B,C,L]    4M
  ushort_t* ADJB= ws + 17039360;                    // [B,L,L]    16M
  ushort_t* SUP = XT;                               // reuse: xt dead after qkv

  // one-time weight transposes + x transpose (bf16)
  cvt_transpose_kernel<<<(CC * HIDN + 255) / 256, 256, 0, stream>>>(wq, WQT, CC, HIDN);
  cvt_transpose_kernel<<<(CC * HIDN + 255) / 256, 256, 0, stream>>>(wk, WKT, CC, HIDN);
  cvt_transpose_kernel<<<(CC * HIDN + 255) / 256, 256, 0, stream>>>(wv, WVT, CC, HIDN);
  cvt_transpose_kernel<<<(HIDN * CC + 255) / 256, 256, 0, stream>>>(wo, WOT, HIDN, CC);
  cvt_transpose_kernel<<<(CC * CC + 255) / 256, 256, 0, stream>>>(weight, WGT, CC, CC);
  cvt_x_kernel<<<(BB * LL * CC) / 256, 256, 0, stream>>>(x, XT);

  qkv_kernel<<<dim3(LL / 16, BB, 3), 256, 0, stream>>>(XT, WQT, WKT, WVT,
                                                       bq, bk, bv, Q, K, VT);
  attn_kernel<<<dim3(LL / 32, BB), 64,
                (2 * 16 * LL) * sizeof(float) + (2 * 16 * LL) * sizeof(ushort_t),
                stream>>>(Q, K, VT, dm, alpha, O, adj);
  owo_kernel<<<dim3(BB * LL / 16, 2), 256, 0, stream>>>(O, WOT, bo, HAT);
  cvt_adj_kernel<<<(BB * LL * LL / 8) / 256, 256, 0, stream>>>(adj, ADJB);
  gcn_support_kernel<<<dim3(LL / 16, BB), 256, 0, stream>>>(ADJB, HAT, SUP);
  out_kernel<<<dim3(LL / 16, BB), 256, 0, stream>>>(SUP, WGT, bias, x, out);
}
